// ASDTransformer_77446850281556
// MI455X (gfx1250) — matrix operand order repaired and checked by reading, not compiled, not run on hardware
//
#include <hip/hip_runtime.h>
#include <hip/hip_bf16.h>

// ---------------------------------------------------------------------------
// MI455X (gfx1250) fused transformer encoder.
//  * all GEMMs through v_wmma_f32_16x16x32_bf16 (bf16 in, fp32 accum)
//  * GEMM tiles staged by the Tensor Data Mover (tensor_load_to_lds),
//    double-buffered in LDS, synchronized with s_wait_tensorcnt
//  * weights pre-transposed to [N][K] at conversion time so no on-path
//    LDS transpose scatter is needed
//  * attention K-tile via TDM (zero-padding 151->160 comes free from the
//    TDM out-of-bounds-reads-return-zero rule)
// ---------------------------------------------------------------------------

typedef __attribute__((ext_vector_type(16))) __bf16 bf16x16;
typedef __attribute__((ext_vector_type(8)))  __bf16 bf16x8;
typedef __attribute__((ext_vector_type(8)))  float  floatx8;
typedef __attribute__((ext_vector_type(4)))  unsigned tdm_g0_t;
typedef __attribute__((ext_vector_type(8)))  int      tdm_g1_t;
typedef __attribute__((ext_vector_type(4)))  int      tdm_g2_t;

#define BATCH 256
#define TLEN  150
#define SLEN  151      // T + 1 (cls)
#define SPAD  160      // padded to 10 tiles of 16
#define EMB   512
#define NHEAD 8
#define DHEAD 64
#define FFDIM 2048
#define NLAYER 6
#define MROWS (BATCH * SLEN)   // 38656 = 64 * 604

__device__ __forceinline__ __bf16 f2bf(float f) {
  unsigned u = __builtin_bit_cast(unsigned, f);
  unsigned r = u + 0x7FFFu + ((u >> 16) & 1u);   // round-to-nearest-even
  unsigned short h = (unsigned short)(r >> 16);
  return __builtin_bit_cast(__bf16, h);
}

__device__ __forceinline__ bf16x16 load_frag2(const __bf16* p0, const __bf16* p1) {
  bf16x8 lo = *(const bf16x8*)p0;
  bf16x8 hi = *(const bf16x8*)p1;
  bf16x16 r;
#pragma unroll
  for (int i = 0; i < 8; ++i) { r[i] = lo[i]; r[i + 8] = hi[i]; }
  return r;
}

__device__ __forceinline__ floatx8 wmma_bf16(bf16x16 a, bf16x16 b, floatx8 c) {
  return __builtin_amdgcn_wmma_f32_16x16x32_bf16(false, a, false, b, (short)0, c,
                                                 false, false);
}

// LDS byte offset of a __shared__ object (flat LDS aperture: addr[31:0])
__device__ __forceinline__ unsigned lds_off(const void* p) {
  return (unsigned)(unsigned long)p;
}

// ---------------------------------------------------------------------------
// TDM 2D tile load: global (row-major, bf16 elements) -> LDS.
// D# packing per CDNA5 ISA.
//   dim0 = contiguous elements, dim1 = rows, stride in elements.
//   rows >= tensor_d1 read as zero (used for seq-length padding).
// ---------------------------------------------------------------------------
__device__ __forceinline__ void tdm_load_2d_bf16(
    unsigned ldsOff, const void* gptr,
    unsigned tensor_d0, unsigned tensor_d1,
    unsigned tile_d0, unsigned tile_d1, unsigned stride_d0) {
  unsigned long ga = (unsigned long)gptr;
  tdm_g0_t g0;
  g0[0] = 1u;                                              // count=1, user desc
  g0[1] = ldsOff;                                          // lds_addr
  g0[2] = (unsigned)ga;                                    // global_addr[31:0]
  g0[3] = ((unsigned)(ga >> 32) & 0x01FFFFFFu)             // global_addr[56:32]
          | 0x80000000u;                                   // type=2 ("image")
  tdm_g1_t g1;
  g1[0] = (int)(1u << 16);                                 // data_size=1 -> 2B
  g1[1] = (int)((tensor_d0 & 0xFFFFu) << 16);              // tensor_dim0[15:0]
  g1[2] = (int)(((tensor_d0 >> 16) & 0xFFFFu) |
                ((tensor_d1 & 0xFFFFu) << 16));            // dim0 hi | dim1 lo
  g1[3] = (int)(((tensor_d1 >> 16) & 0xFFFFu) |
                ((tile_d0 & 0xFFFFu) << 16));              // dim1 hi | tile0
  g1[4] = (int)(tile_d1 & 0xFFFFu);                        // tile1 | tile2=0
  g1[5] = (int)stride_d0;                                  // dim0_stride[31:0]
  g1[6] = 0;
  g1[7] = 0;
  tdm_g2_t z4 = {0, 0, 0, 0};
  tdm_g1_t z8 = {0, 0, 0, 0, 0, 0, 0, 0};
  // clang-23 / therock 6-arg form: (g0, g1, g2, g3, g4, cpol)
  __builtin_amdgcn_tensor_load_to_lds(g0, g1, z4, z4, z8, 0);
}

// ---------------------------------------------------------------------------
// bf16 WMMA GEMM: C[M,N] = A[M,K] @ Wt[N,K]^T + bias, optional relu,
// fp32 or bf16 out. 64x128 block tile, 8 waves of 32x32.
// A and B tiles streamed by TDM into double-buffered LDS.
// M%64==0, N%128==0, K%32==0 (all shapes here qualify).
// ---------------------------------------------------------------------------
__global__ __launch_bounds__(256) void gemm_bf16_wmma(
    const __bf16* __restrict__ A, const __bf16* __restrict__ Wt,
    const float* __restrict__ bias, float* outF, __bf16* outB,
    int M, int N, int K, int doRelu) {
  __shared__ __bf16 As[2][64 * 32];    // A tile  [m][k]
  __shared__ __bf16 Bs[2][128 * 32];   // B tile  [n][k] (weights pre-transposed)

  const int tid  = threadIdx.x;
  const int wave = tid >> 5, lane = tid & 31;
  const int ln   = lane & 15, half = lane >> 4;
  const int wm   = wave >> 2, wn = wave & 3;
  const long m0  = (long)blockIdx.y * 64;
  const int  n0  = blockIdx.x * 128;

  floatx8 acc[2][2];
  acc[0][0] = {}; acc[0][1] = {}; acc[1][0] = {}; acc[1][1] = {};

  const int nk = K >> 5;
  if (wave == 0) {
    tdm_load_2d_bf16(lds_off(As[0]), A + m0 * (long)K, K, 1u << 20, 32, 64, K);
    tdm_load_2d_bf16(lds_off(Bs[0]), Wt + (long)n0 * K, K, 1u << 20, 32, 128, K);
  }

  for (int kt = 0; kt < nk; ++kt) {
    const int cur = kt & 1;
    if (wave == 0) {
      if (kt + 1 < nk) {
        long knext = (long)(kt + 1) * 32;
        tdm_load_2d_bf16(lds_off(As[cur ^ 1]), A + m0 * (long)K + knext,
                         K, 1u << 20, 32, 64, K);
        tdm_load_2d_bf16(lds_off(Bs[cur ^ 1]), Wt + (long)n0 * K + knext,
                         K, 1u << 20, 32, 128, K);
        __builtin_amdgcn_s_wait_tensorcnt(2);  // current tile complete
      } else {
        __builtin_amdgcn_s_wait_tensorcnt(0);
      }
    }
    __syncthreads();

    // A fragments: lane ln = row, runs at half*8 and 16+half*8 (ISA 16-bit A)
    const __bf16* ap0 = &As[cur][(wm * 32 + ln) * 32];
    const __bf16* ap1 = &As[cur][(wm * 32 + 16 + ln) * 32];
    bf16x16 a0 = load_frag2(ap0 + half * 8, ap0 + 16 + half * 8);
    bf16x16 a1 = load_frag2(ap1 + half * 8, ap1 + 16 + half * 8);
    // B fragments: lane ln = col, runs at half*8 and 16+half*8 (same as A)
    const __bf16* bp0 = &Bs[cur][(wn * 32 + ln) * 32];
    const __bf16* bp1 = &Bs[cur][(wn * 32 + 16 + ln) * 32];
    bf16x16 bf0 = load_frag2(bp0 + half * 8, bp0 + 16 + half * 8);
    bf16x16 bf1 = load_frag2(bp1 + half * 8, bp1 + 16 + half * 8);

    acc[0][0] = wmma_bf16(a0, bf0, acc[0][0]);
    acc[0][1] = wmma_bf16(a0, bf1, acc[0][1]);
    acc[1][0] = wmma_bf16(a1, bf0, acc[1][0]);
    acc[1][1] = wmma_bf16(a1, bf1, acc[1][1]);
    __syncthreads();
  }

#pragma unroll
  for (int ms = 0; ms < 2; ++ms)
#pragma unroll
    for (int ns = 0; ns < 2; ++ns) {
      int gcol = n0 + wn * 32 + ns * 16 + ln;
      float bv = bias[gcol];
#pragma unroll
      for (int r = 0; r < 8; ++r) {
        long grow = m0 + wm * 32 + ms * 16 + r + half * 8;  // C layout rows
        float v = acc[ms][ns][r] + bv;
        if (doRelu) v = fmaxf(v, 0.f);
        if (outB) outB[grow * N + gcol] = f2bf(v);
        else      outF[grow * N + gcol] = v;
      }
    }
}

// ---------------------------------------------------------------------------
// Fused attention: one block (128 threads, 4 waves) per (b,h).
// K tile streamed by TDM (rows >= 151 auto-zeroed); V transposed manually;
// scores in registers, shfl softmax, probs staged per-wave to LDS with
// s_wait_dscnt fences; QK^T and P@V through WMMA.
// ---------------------------------------------------------------------------
__global__ __launch_bounds__(128) void attn_kernel(
    const __bf16* __restrict__ qkv,   // [B, SLEN, 3E] bf16
    const int* __restrict__ mask,     // [B, TLEN]
    __bf16* __restrict__ obuf) {      // [B, SLEN, E] bf16
  __shared__ __bf16 kls[SPAD * DHEAD];    // [s][d]
  __shared__ __bf16 vT[DHEAD * SPAD];     // [d][s]
  __shared__ __bf16 pb[4][16 * SPAD];     // per-wave probs [row][col]

  const int b = blockIdx.x >> 3;
  const int h = blockIdx.x & 7;
  const int tid = threadIdx.x, wave = tid >> 5, lane = tid & 31;
  const int ln = lane & 15, half = lane >> 4;
  const __bf16 bz = __builtin_bit_cast(__bf16, (unsigned short)0);

  // K tile via TDM: 64 contiguous elems per row, 151 valid rows, pad to 160
  if (wave == 0)
    tdm_load_2d_bf16(lds_off(kls),
                     qkv + ((long)b * SLEN) * (3 * EMB) + EMB + h * DHEAD,
                     DHEAD, SLEN, DHEAD, SPAD, 3 * EMB);
  // V transposed into LDS manually
  for (int idx = tid; idx < SPAD * DHEAD; idx += 128) {
    int s = idx >> 6, d = idx & 63;
    __bf16 vv = bz;
    if (s < SLEN)
      vv = qkv[((long)b * SLEN + s) * (3 * EMB) + 2 * EMB + h * DHEAD + d];
    vT[d * SPAD + s] = vv;
  }
  if (wave == 0) __builtin_amdgcn_s_wait_tensorcnt(0);
  __syncthreads();

  const float scale = 0.125f;  // 1/sqrt(64)

  for (int qi = wave; qi < 10; qi += 4) {
    bf16x16 a0 = {}, a1 = {};
    int sq = qi * 16 + ln;
    if (sq < SLEN) {
      const __bf16* qp = qkv + ((long)b * SLEN + sq) * (3 * EMB) + h * DHEAD;
      a0 = load_frag2(qp + half * 8, qp + 16 + half * 8);
      a1 = load_frag2(qp + 32 + half * 8, qp + 48 + half * 8);
    }

    floatx8 sacc[10];
#pragma unroll
    for (int j = 0; j < 10; ++j) {
      floatx8 a = {};
      const __bf16* kp = kls + (j * 16 + ln) * DHEAD;
      bf16x16 bk0 = load_frag2(kp + half * 8,      kp + 16 + half * 8);
      bf16x16 bk1 = load_frag2(kp + 32 + half * 8, kp + 48 + half * 8);
      a = wmma_bf16(a0, bk0, a);
      a = wmma_bf16(a1, bk1, a);
      int col = j * 16 + ln;
      float bv;
      if (col == 0)          bv = 0.f;
      else if (col >= SLEN)  bv = -1e9f;
      else                   bv = (mask[b * TLEN + col - 1] == 0) ? -1e9f : 0.f;
#pragma unroll
      for (int r = 0; r < 8; ++r) a[r] = a[r] * scale + bv;
      sacc[j] = a;
    }

    // register softmax per row (rows r + half*8; cols across j and 16 lanes)
#pragma unroll
    for (int r = 0; r < 8; ++r) {
      float m = -3.0e38f;
#pragma unroll
      for (int j = 0; j < 10; ++j) m = fmaxf(m, sacc[j][r]);
      m = fmaxf(m, __shfl_xor(m, 1)); m = fmaxf(m, __shfl_xor(m, 2));
      m = fmaxf(m, __shfl_xor(m, 4)); m = fmaxf(m, __shfl_xor(m, 8));
      float ssum = 0.f;
#pragma unroll
      for (int j = 0; j < 10; ++j) {
        float e = __expf(sacc[j][r] - m);
        sacc[j][r] = e;
        ssum += e;
      }
      ssum += __shfl_xor(ssum, 1); ssum += __shfl_xor(ssum, 2);
      ssum += __shfl_xor(ssum, 4); ssum += __shfl_xor(ssum, 8);
      float inv = 1.f / ssum;
      int row = r + half * 8;
#pragma unroll
      for (int j = 0; j < 10; ++j)
        pb[wave][row * SPAD + j * 16 + ln] = f2bf(sacc[j][r] * inv);
    }
    asm volatile("s_wait_dscnt 0x0" ::: "memory");

    // O(16x64) = P(16x160) @ V(160x64)
#pragma unroll
    for (int ns = 0; ns < 4; ++ns) {
      floatx8 oa = {};
      const __bf16* vp = vT + (ns * 16 + ln) * SPAD;
#pragma unroll
      for (int c = 0; c < 5; ++c) {
        const __bf16* pp = &pb[wave][ln * SPAD + c * 32];
        bf16x16 af = load_frag2(pp + half * 8, pp + 16 + half * 8);
        bf16x16 bf = load_frag2(vp + c * 32 + half * 8,
                                vp + c * 32 + 16 + half * 8);
        oa = wmma_bf16(af, bf, oa);
      }
#pragma unroll
      for (int r = 0; r < 8; ++r) {
        int so = qi * 16 + r + half * 8;
        if (so < SLEN)
          obuf[((long)b * SLEN + so) * EMB + h * DHEAD + ns * 16 + ln] =
              f2bf(oa[r]);
      }
    }
    asm volatile("s_wait_dscnt 0x0" ::: "memory");  // WAR fence before pb reuse
  }
}

// ---------------------------------------------------------------------------
// residual + LayerNorm: one wave per row of 512, writes fp32 x and bf16 copy
// ---------------------------------------------------------------------------
__global__ __launch_bounds__(256) void add_ln_kernel(
    const float* xin, const float* __restrict__ dlt,
    const float* __restrict__ g, const float* __restrict__ bb,
    float* xout, __bf16* __restrict__ xb, int M) {
  int wave = threadIdx.x >> 5, lane = threadIdx.x & 31;
  long row = (long)blockIdx.x * 8 + wave;
  if (row >= M) return;
  const float* xr = xin + row * EMB;
  const float* dr = dlt + row * EMB;
  float v[16], s = 0.f;
#pragma unroll
  for (int i = 0; i < 16; ++i) { v[i] = xr[lane + 32 * i] + dr[lane + 32 * i]; s += v[i]; }
  s += __shfl_xor(s, 1); s += __shfl_xor(s, 2); s += __shfl_xor(s, 4);
  s += __shfl_xor(s, 8); s += __shfl_xor(s, 16);
  float mean = s * (1.f / EMB);
  float vs = 0.f;
#pragma unroll
  for (int i = 0; i < 16; ++i) { float d = v[i] - mean; vs += d * d; }
  vs += __shfl_xor(vs, 1); vs += __shfl_xor(vs, 2); vs += __shfl_xor(vs, 4);
  vs += __shfl_xor(vs, 8); vs += __shfl_xor(vs, 16);
  float rstd = rsqrtf(vs * (1.f / EMB) + 1e-5f);
#pragma unroll
  for (int i = 0; i < 16; ++i) {
    int e = lane + 32 * i;
    float o = (v[i] - mean) * rstd * g[e] + bb[e];
    xout[row * EMB + e] = o;
    xb[row * EMB + e]   = f2bf(o);
  }
}

// ---------------------------------------------------------------------------
// embedding dispatch + pos embed + cls
// ---------------------------------------------------------------------------
__global__ __launch_bounds__(256) void embed_kernel(
    const int* __restrict__ token_type, const int* __restrict__ decision_idx,
    const float* __restrict__ action_val, const float* __restrict__ dis_val,
    const float* __restrict__ v_val, const float* __restrict__ action_W,
    const float* __restrict__ action_b, const float* __restrict__ dis_W,
    const float* __restrict__ dis_b, const float* __restrict__ v_W,
    const float* __restrict__ v_b, const float* __restrict__ dec_emb,
    const float* __restrict__ pos_embed, const float* __restrict__ cls_token,
    float* __restrict__ x, __bf16* __restrict__ xb) {
  long idx = (long)blockIdx.x * 256 + threadIdx.x;
  if (idx >= (long)BATCH * SLEN * EMB) return;
  int e = idx & (EMB - 1);
  long rs = idx >> 9;           // row = b*SLEN + s
  int s = (int)(rs % SLEN);
  int b = (int)(rs / SLEN);
  float v;
  if (s == 0) {
    v = cls_token[e];
  } else {
    int t = s - 1;
    int tok = b * TLEN + t;
    int tt = token_type[tok];
    if (tt == 0) {
      v = action_val[tok] * action_W[e] + action_b[e];
    } else if (tt == 1) {
      v = dis_b[e] + v_b[e];
#pragma unroll
      for (int i = 0; i < 10; ++i) v += dis_val[tok * 10 + i] * dis_W[i * EMB + e];
#pragma unroll
      for (int i = 0; i < 3; ++i)  v += v_val[tok * 3 + i] * v_W[i * EMB + e];
    } else {
      v = dec_emb[decision_idx[tok] * EMB + e];
    }
    v += pos_embed[t * EMB + e];
  }
  x[idx]  = v;
  xb[idx] = f2bf(v);
}

// fp32 [L][K][N] -> bf16 transposed [L][N][K] (weights, once per call)
__global__ __launch_bounds__(256) void cvt_t_kernel(const float* __restrict__ src,
                                                    __bf16* __restrict__ dst,
                                                    int K, int N, long n) {
  long idx = (long)blockIdx.x * 256 + threadIdx.x;
  if (idx >= n) return;
  long kn = (long)K * N;
  int l = (int)(idx / kn);
  long r = idx - (long)l * kn;
  int k = (int)(r / N);
  int nn = (int)(r - (long)k * N);
  dst[(long)l * kn + (long)nn * K + k] = f2bf(src[idx]);
}

__global__ __launch_bounds__(256) void take_cls_kernel(const float* __restrict__ x,
                                                       float* __restrict__ out) {
  int idx = blockIdx.x * 256 + threadIdx.x;
  if (idx >= BATCH * EMB) return;
  int b = idx >> 9, e = idx & (EMB - 1);
  out[idx] = x[(long)b * SLEN * EMB + e];
}

// ---------------------------------------------------------------------------
extern "C" void kernel_launch(void* const* d_in, const int* in_sizes, int n_in,
                              void* d_out, int out_size, void* d_ws, size_t ws_size,
                              hipStream_t stream) {
  (void)in_sizes; (void)n_in; (void)out_size; (void)ws_size;
  const int*   token_type   = (const int*)d_in[0];
  const int*   mask         = (const int*)d_in[1];
  const int*   decision_idx = (const int*)d_in[2];
  const float* action_val   = (const float*)d_in[3];
  const float* dis_val      = (const float*)d_in[4];
  const float* v_val        = (const float*)d_in[5];
  const float* action_W     = (const float*)d_in[6];
  const float* action_b     = (const float*)d_in[7];
  const float* dis_W        = (const float*)d_in[8];
  const float* dis_b        = (const float*)d_in[9];
  const float* v_W          = (const float*)d_in[10];
  const float* v_b          = (const float*)d_in[11];
  const float* dec_emb      = (const float*)d_in[12];
  const float* pos_embed    = (const float*)d_in[13];
  const float* cls_token    = (const float*)d_in[14];
  const float* qkv_W        = (const float*)d_in[15];
  const float* qkv_b        = (const float*)d_in[16];
  const float* out_W        = (const float*)d_in[17];
  const float* out_b        = (const float*)d_in[18];
  const float* ln1_g        = (const float*)d_in[19];
  const float* ln1_b        = (const float*)d_in[20];
  const float* ff1_W        = (const float*)d_in[21];
  const float* ff1_b        = (const float*)d_in[22];
  const float* ff2_W        = (const float*)d_in[23];
  const float* ff2_b        = (const float*)d_in[24];
  const float* ln2_g        = (const float*)d_in[25];
  const float* ln2_b        = (const float*)d_in[26];

  const size_t nWqkv = (size_t)NLAYER * EMB * 3 * EMB;
  const size_t nWout = (size_t)NLAYER * EMB * EMB;
  const size_t nWff  = (size_t)NLAYER * EMB * FFDIM;
  const size_t nX    = (size_t)MROWS * EMB;
  const size_t nH    = (size_t)MROWS * FFDIM;   // also covers MROWS*3E

  char* w = (char*)d_ws;
  size_t off = 0;
  auto take = [&](size_t bytes) -> void* {
    void* p = w + off;
    off = (off + bytes + 255) & ~(size_t)255;
    return p;
  };
  __bf16* wqkvT = (__bf16*)take(nWqkv * 2);   // [L][3E][E]
  __bf16* woutT = (__bf16*)take(nWout * 2);   // [L][E][E]
  __bf16* wff1T = (__bf16*)take(nWff * 2);    // [L][FF][E]
  __bf16* wff2T = (__bf16*)take(nWff * 2);    // [L][E][FF]
  float*  x     = (float*)take(nX * 4);
  __bf16* xb    = (__bf16*)take(nX * 2);
  float*  tmpf  = (float*)take(nX * 4);
  __bf16* obuf  = (__bf16*)take(nX * 2);
  __bf16* big   = (__bf16*)take(nH * 2);  // QKV (bf16) and FF hidden (aliased)

  // weight fp32 -> bf16, transposed to [N][K] (every call; no cached state)
  cvt_t_kernel<<<(int)((nWqkv + 255) / 256), 256, 0, stream>>>(
      qkv_W, wqkvT, EMB, 3 * EMB, (long)nWqkv);
  cvt_t_kernel<<<(int)((nWout + 255) / 256), 256, 0, stream>>>(
      out_W, woutT, EMB, EMB, (long)nWout);
  cvt_t_kernel<<<(int)((nWff + 255) / 256), 256, 0, stream>>>(
      ff1_W, wff1T, EMB, FFDIM, (long)nWff);
  cvt_t_kernel<<<(int)((nWff + 255) / 256), 256, 0, stream>>>(
      ff2_W, wff2T, FFDIM, EMB, (long)nWff);

  // embedding
  embed_kernel<<<(int)(((long)BATCH * SLEN * EMB + 255) / 256), 256, 0, stream>>>(
      token_type, decision_idx, action_val, dis_val, v_val, action_W, action_b,
      dis_W, dis_b, v_W, v_b, dec_emb, pos_embed, cls_token, x, xb);

  const int mb = MROWS / 64;  // 604
  for (int l = 0; l < NLAYER; ++l) {
    // QKV = xb @ Wqkv + b  -> bf16 [M, 3E]
    gemm_bf16_wmma<<<dim3(3 * EMB / 128, mb), 256, 0, stream>>>(
        xb, wqkvT + (size_t)l * EMB * 3 * EMB, qkv_b + (size_t)l * 3 * EMB,
        nullptr, big, MROWS, 3 * EMB, EMB, 0);
    // attention
    attn_kernel<<<BATCH * NHEAD, 128, 0, stream>>>(big, mask, obuf);
    // proj = o @ Wout + b -> fp32
    gemm_bf16_wmma<<<dim3(EMB / 128, mb), 256, 0, stream>>>(
        obuf, woutT + (size_t)l * EMB * EMB, out_b + (size_t)l * EMB,
        tmpf, nullptr, MROWS, EMB, EMB, 0);
    add_ln_kernel<<<MROWS / 8, 256, 0, stream>>>(
        x, tmpf, ln1_g + (size_t)l * EMB, ln1_b + (size_t)l * EMB, x, xb, MROWS);
    // h = relu(xb @ Wff1 + b) -> bf16
    gemm_bf16_wmma<<<dim3(FFDIM / 128, mb), 256, 0, stream>>>(
        xb, wff1T + (size_t)l * EMB * FFDIM, ff1_b + (size_t)l * FFDIM,
        nullptr, big, MROWS, FFDIM, EMB, 1);
    // d = h @ Wff2 + b -> fp32
    gemm_bf16_wmma<<<dim3(EMB / 128, mb), 256, 0, stream>>>(
        big, wff2T + (size_t)l * FFDIM * EMB, ff2_b + (size_t)l * EMB,
        tmpf, nullptr, MROWS, EMB, FFDIM, 0);
    add_ln_kernel<<<MROWS / 8, 256, 0, stream>>>(
        x, tmpf, ln2_g + (size_t)l * EMB, ln2_b + (size_t)l * EMB, x, xb, MROWS);
  }

  take_cls_kernel<<<(BATCH * EMB + 255) / 256, 256, 0, stream>>>(x, (float*)d_out);
}
